// DDIPredictor_10273561772323
// MI455X (gfx1250) — matrix operand order repaired and checked by reading, not compiled, not run on hardware
//
#include <hip/hip_runtime.h>
#include <hip/hip_bf16.h>
#include <math.h>

// ---------------------------------------------------------------------------
// DDI predictor: 2x GCNConv (symmetric-norm) + pair classifier, MI455X gfx1250.
// GEMMs: v_wmma_f32_16x16x32_bf16 (wave32), K-chunk 128, LDS double buffer,
// global_prefetch one chunk ahead.  Scatter-agg: global_atomic_add_f32.
// Bias+ReLU fused into the consumer's f32->bf16 staging.
// ---------------------------------------------------------------------------

#define N_NODES 8192
#define N_EDGES 524288
#define N_PAIRS 262144
#define N_TYPES 86
#define NT_PAD  96   // N_TYPES padded to multiple of 16

typedef __attribute__((ext_vector_type(16))) __bf16 v16bf;
typedef __attribute__((ext_vector_type(8)))  __bf16 v8bf;
typedef __attribute__((ext_vector_type(4)))  __bf16 v4bf;
typedef __attribute__((ext_vector_type(8)))  float  v8f;

__device__ __forceinline__ __bf16 f2bf(float f) {
  union { float f; unsigned u; } in; in.f = f;
  unsigned r = in.u + 0x7FFFu + ((in.u >> 16) & 1u);   // round-to-nearest-even
  union { unsigned short s; __bf16 b; } out; out.s = (unsigned short)(r >> 16);
  return out.b;
}

// ---------------------------------------------------------------------------
// Pre-swizzle a [K x N] f32 weight matrix into bf16 WMMA B-fragment order.
// frag[(k/32)*Npad + n][32]: a lane (col = lane&15, khalf = lane>>4) reads 16
// contiguous bf16 (32B) -> coalesced 512B per 16 lanes.  n >= N zero-padded.
// ---------------------------------------------------------------------------
__global__ void swizzle_b_kernel(const float* __restrict__ W, __bf16* __restrict__ out,
                                 int K, int N, int Npad) {
  int t = blockIdx.x * blockDim.x + threadIdx.x;
  if (t >= K * Npad) return;
  int k = t / Npad, n = t - k * Npad;
  float v = (n < N) ? W[(size_t)k * N + n] : 0.0f;
  out[((size_t)(k >> 5) * Npad + n) * 32 + (k & 31)] = f2bf(v);
}

// ---------------------------------------------------------------------------
// Degree / normalization helpers (deg includes self-loop, so deg >= 1).
// ---------------------------------------------------------------------------
__global__ void deg_init_kernel(float* __restrict__ deg, int n) {
  int i = blockIdx.x * blockDim.x + threadIdx.x;
  if (i < n) deg[i] = 1.0f;                       // self-loop
}
__global__ void deg_acc_kernel(const int* __restrict__ dst, float* __restrict__ deg, int e) {
  int i = blockIdx.x * blockDim.x + threadIdx.x;
  if (i < e) atomicAdd(&deg[dst[i]], 1.0f);
}
__global__ void dinv_kernel(const float* __restrict__ deg, float* __restrict__ dinv, int n) {
  int i = blockIdx.x * blockDim.x + threadIdx.x;
  if (i < n) dinv[i] = rsqrtf(deg[i]);
}

// h[i,f] = y[i,f] / deg[i]   (self-loop contribution: dinv[i]^2 = 1/deg[i])
__global__ void agg_self_kernel(const float* __restrict__ y, const float* __restrict__ deg,
                                float* __restrict__ h, int F, int total) {
  int t = blockIdx.x * blockDim.x + threadIdx.x;
  if (t < total) h[t] = y[t] / deg[t / F];
}

// scatter-add: h[dst,f] += y[src,f] * dinv[src]*dinv[dst]
__global__ void agg_edges_kernel(const float* __restrict__ y, const int* __restrict__ src,
                                 const int* __restrict__ dst, const float* __restrict__ dinv,
                                 float* __restrict__ h, int F, int E) {
  int tid = threadIdx.x;
  int epb = blockDim.x / F;                       // edges per block
  int e = blockIdx.x * epb + tid / F;
  int f = tid & (F - 1);                          // F is 64 or 128
  if (e < E) {
    int s = src[e], d = dst[e];
    float c = dinv[s] * dinv[d];
    atomicAdd(&h[(size_t)d * F + f], y[(size_t)s * F + f] * c);
  }
}

// ---------------------------------------------------------------------------
// Stage one 16x128 A tile (f32 -> bf16) into LDS, optional fused relu(a+bias),
// float4 global loads + prefetch of the chunk after next.
// ---------------------------------------------------------------------------
__device__ __forceinline__ void stage_tile(const float* __restrict__ A,
                                           const float* __restrict__ bias,
                                           __bf16* __restrict__ dstb,
                                           int m0, int K, int k0,
                                           int tid, int nthreads) {
  for (int idx = tid * 4; idx < 16 * 128; idx += nthreads * 4) {
    int r = idx >> 7, kk = idx & 127;
    const float* p = A + (size_t)(m0 + r) * K + (k0 + kk);
    if ((kk & 31) == 0 && (k0 + 128) < K)
      __builtin_prefetch(p + 128, 0, 1);          // global_prefetch_b8, next chunk
    const float4 v4 = *(const float4*)p;
    float vv[4] = {v4.x, v4.y, v4.z, v4.w};
    v4bf o;
#pragma unroll
    for (int i = 0; i < 4; ++i) {
      float v = vv[i];
      if (bias) { v += bias[k0 + kk + i]; v = v > 0.0f ? v : 0.0f; }
      o[i] = f2bf(v);
    }
    *(v4bf*)(dstb + idx) = o;
  }
}

// ---------------------------------------------------------------------------
// Generic GEMM  Y[M x N] = act(A)[M x K] (f32 row-major) * Bs (swizzled bf16).
// Grid.x = M/16; block = (N/16) waves * 32; each wave owns one 16-col tile.
// K chunked by 128 with LDS double buffering; 4 WMMAs per chunk per wave.
// If bias != nullptr, A is read as relu(A + bias[k]) (fused GCN activation).
// ---------------------------------------------------------------------------
__global__ void gemm_bf16_wmma_kernel(const float* __restrict__ A,
                                      const __bf16* __restrict__ Bs,
                                      const float* __restrict__ bias,
                                      float* __restrict__ Y, int K, int N) {
  __shared__ alignas(16) __bf16 As[2][16 * 128];
  const int m0   = blockIdx.x * 16;
  const int tid  = threadIdx.x;
  const int lane = tid & 31;
  const int n0   = (tid >> 5) * 16;               // wave's N tile
  const int ln   = lane & 15;
  const int half = lane >> 4;
  const int nc   = K >> 7;                        // #128-wide chunks
  v8f acc = {};

  stage_tile(A, bias, As[0], m0, K, 0, tid, blockDim.x);
  __syncthreads();
  int buf = 0;
  for (int c = 0; c < nc; ++c) {
    if (c + 1 < nc)                               // overlap: stage next chunk
      stage_tile(A, bias, As[buf ^ 1], m0, K, (c + 1) << 7, tid, blockDim.x);
#pragma unroll
    for (int kk = 0; kk < 128; kk += 32) {
      // A fragment: row = ln; K-halves contiguous {0-15} / {16-31} (matches B)
      const __bf16* ap = As[buf] + ln * 128 + kk;
      v8bf alo = *(const v8bf*)(ap + half * 16);
      v8bf ahi = *(const v8bf*)(ap + half * 16 + 8);
      v16bf a = __builtin_shufflevector(alo, ahi, 0,1,2,3,4,5,6,7,8,9,10,11,12,13,14,15);
      const int kb = (c << 7) + kk;
      const __bf16* bp = Bs + ((size_t)(kb >> 5) * N + (n0 + ln)) * 32 + (half << 4);
      v16bf b = *(const v16bf*)bp;
      acc = __builtin_amdgcn_wmma_f32_16x16x32_bf16(false, a, false, b, (short)0, acc,
                                                    false, false);
    }
    __syncthreads();
    buf ^= 1;
  }
  const int mb = half * 8;                        // C/D layout: M = r + 8*(lane>=16)
#pragma unroll
  for (int r = 0; r < 8; ++r)
    Y[(size_t)(m0 + mb + r) * N + (n0 + ln)] = acc[r];
}

// ---------------------------------------------------------------------------
// Pair classifier: out[p,t] = sigmoid( [g(d1),g(d2)] @ Wfc + bfc ),
// g(d) = relu(h2raw[d] + b2) applied during the gather.  16 pairs per WG;
// gathered 16x128 bf16 A tile in LDS; 6 waves cover N padded to 96;
// bias + sigmoid fused; columns >= 86 dropped on store.
// ---------------------------------------------------------------------------
__global__ void pair_fc_kernel(const float* __restrict__ h2raw,
                               const float* __restrict__ b2,
                               const int* __restrict__ d1, const int* __restrict__ d2,
                               const __bf16* __restrict__ Wfcs,
                               const float* __restrict__ bfc,
                               float* __restrict__ out) {
  __shared__ alignas(16) __bf16 As[16 * 128];
  const int p0   = blockIdx.x * 16;
  const int tid  = threadIdx.x;
  const int lane = tid & 31;
  const int n0   = (tid >> 5) * 16;
  const int ln   = lane & 15;
  const int half = lane >> 4;
  // gather + fused relu(+b2) + convert; each float4 stays inside one drug half
  for (int i4 = tid; i4 < 512; i4 += blockDim.x) {
    int idx = i4 * 4;
    int r = idx >> 7, f = idx & 127;
    int node = (f < 64) ? d1[p0 + r] : d2[p0 + r];
    int fb = f & 63;
    const float4 v4 = *(const float4*)(h2raw + (size_t)node * 64 + fb);
    float vv[4] = {v4.x, v4.y, v4.z, v4.w};
    v4bf o;
#pragma unroll
    for (int i = 0; i < 4; ++i) {
      float v = vv[i] + b2[fb + i];
      v = v > 0.0f ? v : 0.0f;
      o[i] = f2bf(v);
    }
    *(v4bf*)(As + idx) = o;
  }
  __syncthreads();
  v8f acc = {};
#pragma unroll
  for (int k0 = 0; k0 < 128; k0 += 32) {
    const __bf16* ap = As + ln * 128 + k0;
    v8bf alo = *(const v8bf*)(ap + half * 16);
    v8bf ahi = *(const v8bf*)(ap + half * 16 + 8);
    v16bf a = __builtin_shufflevector(alo, ahi, 0,1,2,3,4,5,6,7,8,9,10,11,12,13,14,15);
    const __bf16* bp = Wfcs + ((size_t)(k0 >> 5) * NT_PAD + (n0 + ln)) * 32 + (half << 4);
    v16bf b = *(const v16bf*)bp;
    acc = __builtin_amdgcn_wmma_f32_16x16x32_bf16(false, a, false, b, (short)0, acc,
                                                  false, false);
  }
  const int n = n0 + ln;
  if (n < N_TYPES) {
    const float bb = bfc[n];
    const int mb = half * 8;
#pragma unroll
    for (int r = 0; r < 8; ++r) {
      float v = acc[r] + bb;
      out[(size_t)(p0 + mb + r) * N_TYPES + n] = 1.0f / (1.0f + __expf(-v));
    }
  }
}

// ---------------------------------------------------------------------------
// Workspace layout (bytes); total ~14.2 MB.
// ---------------------------------------------------------------------------
#define OFF_Y1   0u                                  // 8192*128*4 = 4 MiB
#define OFF_H1   (OFF_Y1  + 8192u*128u*4u)
#define OFF_Y2   (OFF_H1  + 8192u*128u*4u)           // 8192*64*4 = 2 MiB
#define OFF_H2   (OFF_Y2  + 8192u*64u*4u)
#define OFF_DEG  (OFF_H2  + 8192u*64u*4u)            // 32 KiB
#define OFF_DINV (OFF_DEG + 8192u*4u)
#define OFF_W1S  (OFF_DINV+ 8192u*4u)                // 8192*128*2 = 2 MiB
#define OFF_W2S  (OFF_W1S + 8192u*128u*2u)           // 128*64*2
#define OFF_WFCS (OFF_W2S + 128u*64u*2u)             // 128*96*2
#define OFF_END  (OFF_WFCS+ 128u*96u*2u)

extern "C" void kernel_launch(void* const* d_in, const int* in_sizes, int n_in,
                              void* d_out, int out_size, void* d_ws, size_t ws_size,
                              hipStream_t stream) {
  const float* x   = (const float*)d_in[0];
  const int*   ei  = (const int*)d_in[1];
  const int*   src = ei;
  const int*   dst = ei + N_EDGES;
  const int*   d1  = (const int*)d_in[2];
  const int*   d2  = (const int*)d_in[3];
  const float* W1  = (const float*)d_in[4];
  const float* b1  = (const float*)d_in[5];
  const float* W2  = (const float*)d_in[6];
  const float* b2  = (const float*)d_in[7];
  const float* Wfc = (const float*)d_in[8];
  const float* bfc = (const float*)d_in[9];
  float* out = (float*)d_out;

  char* ws = (char*)d_ws;
  float*  y1   = (float*)(ws + OFF_Y1);
  float*  h1   = (float*)(ws + OFF_H1);
  float*  y2   = (float*)(ws + OFF_Y2);
  float*  h2   = (float*)(ws + OFF_H2);
  float*  deg  = (float*)(ws + OFF_DEG);
  float*  dinv = (float*)(ws + OFF_DINV);
  __bf16* W1s  = (__bf16*)(ws + OFF_W1S);
  __bf16* W2s  = (__bf16*)(ws + OFF_W2S);
  __bf16* Wfcs = (__bf16*)(ws + OFF_WFCS);

  // --- weight pre-swizzle (bf16 fragment order) ---
  swizzle_b_kernel<<<(8192*128 + 255)/256, 256, 0, stream>>>(W1,  W1s,  8192, 128, 128);
  swizzle_b_kernel<<<(128*64   + 255)/256, 256, 0, stream>>>(W2,  W2s,  128,  64,  64);
  swizzle_b_kernel<<<(128*NT_PAD+255)/256, 256, 0, stream>>>(Wfc, Wfcs, 128, N_TYPES, NT_PAD);

  // --- degree + normalization ---
  deg_init_kernel<<<N_NODES/256, 256, 0, stream>>>(deg, N_NODES);
  deg_acc_kernel<<<N_EDGES/256, 256, 0, stream>>>(dst, deg, N_EDGES);
  dinv_kernel<<<N_NODES/256, 256, 0, stream>>>(deg, dinv, N_NODES);

  // --- layer 1: y1 = x @ W1 ; GCN aggregate (bias/relu deferred to consumer) ---
  gemm_bf16_wmma_kernel<<<N_NODES/16, 256, 0, stream>>>(x, W1s, nullptr, y1, 8192, 128);
  agg_self_kernel<<<(N_NODES*128)/256, 256, 0, stream>>>(y1, deg, h1, 128, N_NODES*128);
  agg_edges_kernel<<<N_EDGES/2, 256, 0, stream>>>(y1, src, dst, dinv, h1, 128, N_EDGES);

  // --- layer 2: y2 = relu(h1 + b1) @ W2 (bias fused in A staging) ; aggregate ---
  gemm_bf16_wmma_kernel<<<N_NODES/16, 128, 0, stream>>>(h1, W2s, b1, y2, 128, 64);
  agg_self_kernel<<<(N_NODES*64)/256, 256, 0, stream>>>(y2, deg, h2, 64, N_NODES*64);
  agg_edges_kernel<<<N_EDGES/4, 256, 0, stream>>>(y2, src, dst, dinv, h2, 64, N_EDGES);

  // --- pair classifier (relu(h2+b2) fused in gather) ---
  pair_fc_kernel<<<N_PAIRS/16, (NT_PAD/16)*32, 0, stream>>>(h2, b2, d1, d2, Wfcs, bfc, out);
}
